// GCNFraudDetector_63685775065301
// MI455X (gfx1250) — matrix operand order repaired and checked by reading, not compiled, not run on hardware
//
#include <hip/hip_runtime.h>
#include <hip/hip_bf16.h>
#include <cstdint>
#include <cstddef>

#define NODES_C 100000
#define IN_C 256
#define HID 128

typedef __attribute__((ext_vector_type(16))) __bf16 bf16x16;
typedef __attribute__((ext_vector_type(8)))  float  f32x8;

static __device__ __forceinline__ unsigned short f2bf(float f) {
    union { float f; uint32_t u; } v; v.f = f;
    uint32_t u = v.u;
    u += 0x7FFFu + ((u >> 16) & 1u);   // round-to-nearest-even
    return (unsigned short)(u >> 16);
}

// ---------------------------------------------------------------- elementwise
__global__ void fill_f32(float* p, float val, int n) {
    int i = blockIdx.x * blockDim.x + threadIdx.x;
    if (i < n) p[i] = val;
}

__global__ void degree_kernel(const long long* ei, int E, float* deg) {
    int e = blockIdx.x * blockDim.x + threadIdx.x;
    if (e < E) atomicAdd(&deg[(int)ei[E + e]], 1.0f);   // dst row of edge_index
}

__global__ void rsqrt_kernel(float* p, int n) {
    int i = blockIdx.x * blockDim.x + threadIdx.x;
    if (i < n) { float d = p[i]; p[i] = d > 0.0f ? rsqrtf(d) : 0.0f; }
}

__global__ void f32_to_bf16_x4(const float* __restrict__ src,
                               unsigned short* __restrict__ dst, int n4) {
    int i = blockIdx.x * blockDim.x + threadIdx.x;
    if (i < n4) {
        float4 v = ((const float4*)src)[i];
        ushort4 o;
        o.x = f2bf(v.x); o.y = f2bf(v.y); o.z = f2bf(v.z); o.w = f2bf(v.w);
        ((ushort4*)dst)[i] = o;
    }
}

// W [K, Nc] (row-major) -> WT [Nc, K] bf16
__global__ void transpose_w_bf16(const float* __restrict__ W,
                                 unsigned short* __restrict__ WT, int K, int Nc) {
    int i = blockIdx.x * blockDim.x + threadIdx.x;
    if (i < K * Nc) {
        int k = i / Nc, c = i % Nc;
        WT[c * K + k] = f2bf(W[i]);
    }
}

__global__ void bias_relu_bf16(const float* __restrict__ agg,
                               const float* __restrict__ bias,
                               unsigned short* __restrict__ out, int n, int Cmask) {
    int i = blockIdx.x * blockDim.x + threadIdx.x;
    if (i < n) {
        float v = agg[i] + bias[i & Cmask];
        out[i] = f2bf(v > 0.0f ? v : 0.0f);
    }
}

// ---------------------------------------------------------------- WMMA GEMM
// C[Mpad, NCOLS] = A[Mpad, KDIM](bf16) @ B[KDIM, NCOLS], BT[NCOLS, KDIM](bf16)
// staged in LDS.  8 waves/block, each wave computes MT 16-row tiles x NCOLS.
// Mpad is a multiple of 8*16*MT (caller pads buffers) -> NO divergent guards,
// EXEC is statically all-ones around every WMMA.
template <int KDIM, int NCOLS, int MT>
__global__ __launch_bounds__(256) void wmma_gemm_bf16(
        const unsigned short* __restrict__ A,
        const unsigned short* __restrict__ BT,
        float* __restrict__ C) {
    constexpr int NFRAG = NCOLS / 16;
    constexpr int KCH   = KDIM / 32;
    __shared__ unsigned short Bs[NCOLS * KDIM];        // 64KB (L1) / 16KB (L2)

    // cooperative stage of BT -> LDS, 16B chunks
    constexpr int CH16 = NCOLS * KDIM / 8;
    for (int i = threadIdx.x; i < CH16; i += 256)
        ((uint4*)Bs)[i] = ((const uint4*)BT)[i];
    __syncthreads();

    const int wave = threadIdx.x >> 5;
    const int lane = threadIdx.x & 31;
    const int r    = lane & 15;
    const int half = lane >> 4;
    const int rowBase = (blockIdx.x * 8 + wave) * (16 * MT);

    const unsigned short* arow[MT];
#pragma unroll
    for (int m = 0; m < MT; ++m)
        arow[m] = A + (size_t)(rowBase + m * 16 + r) * KDIM;

    f32x8 acc[MT][NFRAG];
#pragma unroll
    for (int m = 0; m < MT; ++m)
#pragma unroll
        for (int f = 0; f < NFRAG; ++f) acc[m][f] = f32x8{};

#pragma unroll
    for (int kc = 0; kc < KCH; ++kc) {
        const int k0 = kc * 32;
        // 16-bit A 16x32 layout: lane row = lane&15, K = elem + 16*half (contiguous)
        union { uint4 q[2]; bf16x16 v; } af[MT];
#pragma unroll
        for (int m = 0; m < MT; ++m) {
            af[m].q[0] = *(const uint4*)(arow[m] + k0 + 16 * half);
            af[m].q[1] = *(const uint4*)(arow[m] + k0 + 16 * half + 8);
        }
#pragma unroll
        for (int f = 0; f < NFRAG; ++f) {
            // 16-bit B 32x16 layout: lane col = lane&15, K = elem + 16*half (contiguous)
            const unsigned short* bcol = Bs + (f * 16 + r) * KDIM + k0 + 16 * half;
            union { uint4 q[2]; bf16x16 v; } bf_;
            bf_.q[0] = *(const uint4*)(bcol);
            bf_.q[1] = *(const uint4*)(bcol + 8);
#pragma unroll
            for (int m = 0; m < MT; ++m) {
                acc[m][f] = __builtin_amdgcn_wmma_f32_16x16x32_bf16(
                    false, af[m].v, false, bf_.v, (short)0, acc[m][f], false, false);
            }
        }
    }
    // C/D layout: VGPR rr -> row rr + 8*half, col = lane&15
#pragma unroll
    for (int m = 0; m < MT; ++m) {
#pragma unroll
        for (int f = 0; f < NFRAG; ++f) {
#pragma unroll
            for (int rr = 0; rr < 8; ++rr) {
                C[(size_t)(rowBase + m * 16 + rr + 8 * half) * NCOLS + f * 16 + r]
                    = acc[m][f][rr];
            }
        }
    }
}

// ---------------------------------------------------------------- scatter-add
// One wave per (edge or self-loop). msg = h[src]*dinv[src]*dinv[dst] -> agg[dst]
template <int C>
__global__ __launch_bounds__(256) void scatter_kernel(
        const float* __restrict__ h, const long long* __restrict__ ei,
        const float* __restrict__ dinv, float* __restrict__ agg, int E, int Nn) {
    constexpr int VW = C / 32;  // floats per lane (4 or 2)
    const int gw   = blockIdx.x * (blockDim.x >> 5) + (threadIdx.x >> 5);
    const int lane = threadIdx.x & 31;
    const int ET = E + Nn;
    if (gw >= ET) return;
    int s, d; float nrm;
    if (gw < E) {
        s = (int)ei[gw];          // src row
        d = (int)ei[E + gw];      // dst row
        nrm = dinv[s] * dinv[d];
    } else {
        s = d = gw - E;           // self loop
        float di = dinv[s];
        nrm = di * di;
    }
    const float* hp = h + (size_t)s * C + lane * VW;
    float* ap = agg + (size_t)d * C + lane * VW;
#pragma unroll
    for (int j = 0; j < VW; ++j) atomicAdd(&ap[j], hp[j] * nrm);
}

// ---------------------------------------------------------------- head
// wave per node: relu(agg2 + b2) @ Wfc + bfc, 2-class log_softmax
__global__ __launch_bounds__(256) void head_kernel(
        const float* __restrict__ agg, const float* __restrict__ b2,
        const float* __restrict__ Wfc, const float* __restrict__ bfc,
        float* __restrict__ out, int Nn) {
    const int gw   = blockIdx.x * 8 + (threadIdx.x >> 5);
    const int lane = threadIdx.x & 31;
    if (gw >= Nn) return;
    const float* row = agg + (size_t)gw * 64;
    float p0 = 0.0f, p1 = 0.0f;
#pragma unroll
    for (int j = 0; j < 2; ++j) {
        int c = lane + j * 32;
        float v = row[c] + b2[c];
        v = v > 0.0f ? v : 0.0f;
        p0 += v * Wfc[c * 2 + 0];
        p1 += v * Wfc[c * 2 + 1];
    }
#pragma unroll
    for (int off = 16; off > 0; off >>= 1) {
        p0 += __shfl_down(p0, off, 32);
        p1 += __shfl_down(p1, off, 32);
    }
    if (lane == 0) {
        float l0 = p0 + bfc[0], l1 = p1 + bfc[1];
        float m  = fmaxf(l0, l1);
        float lse = m + __logf(__expf(l0 - m) + __expf(l1 - m));
        out[gw * 2 + 0] = l0 - lse;
        out[gw * 2 + 1] = l1 - lse;
    }
}

// ---------------------------------------------------------------- launcher
extern "C" void kernel_launch(void* const* d_in, const int* in_sizes, int n_in,
                              void* d_out, int out_size, void* d_ws, size_t ws_size,
                              hipStream_t stream) {
    const float*     x   = (const float*)d_in[0];
    const long long* ei  = (const long long*)d_in[1];
    const float*     W1  = (const float*)d_in[2];
    const float*     b1  = (const float*)d_in[3];
    const float*     W2  = (const float*)d_in[4];
    const float*     b2  = (const float*)d_in[5];
    const float*     Wfc = (const float*)d_in[6];
    const float*     bfc = (const float*)d_in[7];
    float* out = (float*)d_out;

    const int Nn = in_sizes[0] / IN_C;   // 100000
    const int E  = in_sizes[1] / 2;      // 1600000
    (void)n_in; (void)out_size; (void)ws_size;

    const int ROWS_PER_BLOCK = 8 * 16 * 2;              // 8 waves x 2 tiles x 16 rows
    const int Npad = ((Nn + ROWS_PER_BLOCK - 1) / ROWS_PER_BLOCK) * ROWS_PER_BLOCK;
    const int GEMM_BLOCKS = Npad / ROWS_PER_BLOCK;

    // workspace carve-out (region reuse); M-padded buffers so the WMMA GEMM
    // needs no bounds checks (pad rows compute garbage that is never read).
    char* ws = (char*)d_ws;
    size_t off = 0;
    auto alloc = [&](size_t bytes) {
        size_t o = off; off = (off + bytes + 255) & ~(size_t)255; return o;
    };
    float*          dinv = (float*)(ws + alloc((size_t)Nn * 4));
    unsigned short* W1T  = (unsigned short*)(ws + alloc((size_t)IN_C * HID * 2));
    unsigned short* W2T  = (unsigned short*)(ws + alloc((size_t)HID * (HID / 2) * 2));
    char* regionA = ws + alloc((size_t)Npad * IN_C * 2);            // xb, then a1+h2
    float* h1     = (float*)(ws + alloc((size_t)Npad * HID * 4));
    float* agg1   = (float*)(ws + alloc((size_t)Nn * HID * 4));
    unsigned short* xb = (unsigned short*)regionA;                  // [Npad, 256] bf16
    unsigned short* a1 = (unsigned short*)regionA;                  // [Npad, 128] bf16
    float* h2   = (float*)(regionA + (size_t)Npad * HID * 2);       // [Npad, 64] f32
    float* agg2 = agg1;                                             // reuse

    const int T = 256;
    const int ET = E + Nn;

    // ---- gcn_norm: deg (self-loop = 1), atomic count, rsqrt
    fill_f32<<<(Nn + T - 1) / T, T, 0, stream>>>(dinv, 1.0f, Nn);
    degree_kernel<<<(E + T - 1) / T, T, 0, stream>>>(ei, E, dinv);
    rsqrt_kernel<<<(Nn + T - 1) / T, T, 0, stream>>>(dinv, Nn);

    // ---- layer 1: h1 = x @ W1 (WMMA bf16 + LDS-staged B), scatter, bias+relu
    int n4 = Nn * IN_C / 4;
    f32_to_bf16_x4<<<(n4 + T - 1) / T, T, 0, stream>>>(x, xb, n4);
    int w1n = IN_C * HID;
    transpose_w_bf16<<<(w1n + T - 1) / T, T, 0, stream>>>(W1, W1T, IN_C, HID);
    int agg1n = Nn * HID;
    fill_f32<<<(agg1n + T - 1) / T, T, 0, stream>>>(agg1, 0.0f, agg1n);
    wmma_gemm_bf16<IN_C, HID, 2><<<GEMM_BLOCKS, T, 0, stream>>>(xb, W1T, h1);
    scatter_kernel<HID><<<(ET + 7) / 8, T, 0, stream>>>(h1, ei, dinv, agg1, E, Nn);
    bias_relu_bf16<<<(agg1n + T - 1) / T, T, 0, stream>>>(agg1, b1, a1, agg1n, HID - 1);

    // ---- layer 2: h2 = a1 @ W2 (WMMA bf16 + LDS-staged B), scatter
    int w2n = HID * (HID / 2);
    transpose_w_bf16<<<(w2n + T - 1) / T, T, 0, stream>>>(W2, W2T, HID, HID / 2);
    int agg2n = Nn * (HID / 2);
    fill_f32<<<(agg2n + T - 1) / T, T, 0, stream>>>(agg2, 0.0f, agg2n);
    wmma_gemm_bf16<HID, HID / 2, 2><<<GEMM_BLOCKS, T, 0, stream>>>(a1, W2T, h2);
    scatter_kernel<HID / 2><<<(ET + 7) / 8, T, 0, stream>>>(h2, ei, dinv, agg2, E, Nn);

    // ---- head: fc + log_softmax
    head_kernel<<<(Nn + 7) / 8, T, 0, stream>>>(agg2, b2, Wfc, bfc, out, Nn);
}
